// DecomposeModel_12867722018978
// MI455X (gfx1250) — compile-verified, matrix operand order checked by reading, not run on hardware
//
#include <hip/hip_runtime.h>
#include <hip/hip_bf16.h>
#include <math.h>

// ---------------------------------------------------------------------------
// CDNA5 (gfx1250) bf16-WMMA fused gated-MLP.
//   out = tanh(A@Wh + bh) * sigmoid(A@Wg + bg)        (4 GEMM-gate passes)
// - weights pre-transposed/converted to bf16 [N][K]; intermediates bf16
// - 128x64 block tile, 8 waves; per wave 2Mx2N x {h,g} = 8 WMMA per K-step
// - bf16 A (layer 2) and B tiles staged with GLOBAL_LOAD_ASYNC_TO_LDS_B128
//   into double-buffered LDS (ASYNCcnt); f32 A (layer 1) staged via registers
//   with native float->bf16 conversion
// ---------------------------------------------------------------------------

typedef __attribute__((ext_vector_type(16))) __bf16        v16bf;
typedef __attribute__((ext_vector_type(2)))  __bf16        v2bf;
typedef __attribute__((ext_vector_type(8)))  float         v8f;
typedef __attribute__((ext_vector_type(4)))  unsigned int  v4u;
typedef __attribute__((ext_vector_type(8)))  unsigned int  v8u;
typedef __attribute__((ext_vector_type(4)))  int           v4i;

#define BM   128
#define BN   64
#define BKK  32
#define LDK  40          // padded K stride (ushorts): b128-aligned, conflict-free
#define HDIM 512

#define AS1 __attribute__((address_space(1)))
#define AS3 __attribute__((address_space(3)))

#if __has_builtin(__builtin_amdgcn_global_load_async_to_lds_b128) && \
    __has_builtin(__builtin_amdgcn_s_wait_asynccnt)
#define USE_ASYNC 1
#else
#define USE_ASYNC 0
#endif

__device__ __forceinline__ unsigned short f2bf(float f) {
  __bf16 h = (__bf16)f;                         // native cvt if target has it
  return __builtin_bit_cast(unsigned short, h);
}

__device__ __forceinline__ unsigned int f2bf_pk(float a, float b) {
#if __has_builtin(__builtin_amdgcn_cvt_pk_bf16_f32)
  v2bf p = __builtin_amdgcn_cvt_pk_bf16_f32(a, b);
#else
  v2bf p; p.x = (__bf16)a; p.y = (__bf16)b;     // lowers to v_cvt_pk_bf16_f32 when available
#endif
  return __builtin_bit_cast(unsigned int, p);
}

__device__ __forceinline__ float tanh_fast(float x) {
  float xc = fminf(fmaxf(x, -15.0f), 15.0f);
  float e  = __expf(2.0f * xc);                  // v_exp_f32, branch-free
  return __fdividef(e - 1.0f, e + 1.0f);
}

__device__ __forceinline__ float sigmoid_fast(float x) {
  return __fdividef(1.0f, 1.0f + __expf(-x));
}

union Frag {
  v8u   u;
  v16bf v;
};

__device__ __forceinline__ v8f wmma_bf16(const Frag& a, const Frag& b, v8f c) {
  return __builtin_amdgcn_wmma_f32_16x16x32_bf16(false, a.v, false, b.v, (short)0, c, false, false);
}

// A pieces: f32 (ABF16=false) or pre-converted bf16 (ABF16=true); each 512 wide.
// Wh/Wg: bf16, transposed [512 N][Ktot K], K contiguous.
template <bool ABF16, bool OUTBF16>
__global__ __launch_bounds__(256)
void gate_gemm(const void* __restrict__ A0v,
               const void* __restrict__ A1v,
               const void* __restrict__ A2v,
               int Ktot,
               const unsigned short* __restrict__ Wh, const float* __restrict__ bh,
               const unsigned short* __restrict__ Wg, const float* __restrict__ bg,
               void* __restrict__ outv,
               const int* __restrict__ group)   // non-null => fuse select; outv holds left_out (f32)
{
  constexpr bool AASYNC = (ABF16 && USE_ASYNC);  // A fully async (bf16 copy)

  __shared__ __align__(16) unsigned short sA [2][BM * LDK];
  __shared__ __align__(16) unsigned short sBh[2][BN * LDK];
  __shared__ __align__(16) unsigned short sBg[2][BN * LDK];

  const int tid   = threadIdx.x;
  const int lane  = tid & 31;
  const int wid   = tid >> 5;          // 8 waves: 4 (M) x 2 (N)
  const int laneN = lane & 15;
  const int half  = lane >> 4;
  const int waveM = wid & 3;
  const int waveN = wid >> 2;

  const int mbase = blockIdx.y * BM;
  const int nbase = blockIdx.x * BN;

  v8f zero = {0.f,0.f,0.f,0.f,0.f,0.f,0.f,0.f};
  v8f ah00 = zero, ah01 = zero, ah10 = zero, ah11 = zero;
  v8f ag00 = zero, ag01 = zero, ag10 = zero, ag11 = zero;

  float4 raF[4];       // staging regs (f32 A mode)
  uint4  raB[2];       // staging regs (bf16 A, no-async fallback)
  uint4  rbh, rbg;     // staging regs (B fallback path)

  const int br = tid >> 2, bc = tid & 3;     // B tile: 64 N-rows x 4 uint4

  auto pieceA = [&](int kb) -> const void* {
    const int piece = kb >> 9;
    return (piece == 0) ? A0v : (piece == 1) ? A1v : A2v;
  };

  auto fetchA = [&](int kb, int buf) {
    const int koff = kb & 511;
    if constexpr (ABF16) {
      const unsigned short* As = (const unsigned short*)pieceA(kb);
#if USE_ASYNC
      #pragma unroll
      for (int it = 0; it < 2; ++it) {
        int idx = tid + it * 256;            // 512 uint4: 128 rows x 4
        int r = idx >> 2, c = idx & 3;
        __builtin_amdgcn_global_load_async_to_lds_b128(
            (AS1 v4i*)(AS1 void*)(As + (size_t)(mbase + r) * HDIM + koff + c * 8),
            (AS3 v4i*)(AS3 void*)&sA[buf][r * LDK + c * 8], 0, 0);
      }
#else
      (void)buf;
      #pragma unroll
      for (int it = 0; it < 2; ++it) {
        int idx = tid + it * 256;
        int r = idx >> 2, c = idx & 3;
        raB[it] = *(const uint4*)(As + (size_t)(mbase + r) * HDIM + koff + c * 8);
      }
#endif
    } else {
      (void)buf;
      const float* As = (const float*)pieceA(kb);
      #pragma unroll
      for (int it = 0; it < 4; ++it) {
        int idx = tid + it * 256;            // 1024 float4: 128 rows x 8
        int r = idx >> 3, c4 = idx & 7;
        raF[it] = *(const float4*)(As + (size_t)(mbase + r) * HDIM + koff + c4 * 4);
      }
    }
  };

  auto fetchB = [&](int kb, int buf) {
    size_t wo = (size_t)(nbase + br) * Ktot + kb + bc * 8;
#if USE_ASYNC
    __builtin_amdgcn_global_load_async_to_lds_b128(
        (AS1 v4i*)(AS1 void*)(Wh + wo),
        (AS3 v4i*)(AS3 void*)&sBh[buf][br * LDK + bc * 8], 0, 0);
    __builtin_amdgcn_global_load_async_to_lds_b128(
        (AS1 v4i*)(AS1 void*)(Wg + wo),
        (AS3 v4i*)(AS3 void*)&sBg[buf][br * LDK + bc * 8], 0, 0);
#else
    (void)buf;
    rbh = *(const uint4*)(Wh + wo);
    rbg = *(const uint4*)(Wg + wo);
#endif
  };

  auto storeA = [&]() {                        // used only when !AASYNC
    if constexpr (ABF16) {
      #pragma unroll
      for (int it = 0; it < 2; ++it) {
        int idx = tid + it * 256;
        int r = idx >> 2, c = idx & 3;
        *(uint4*)&sA[0][r * LDK + c * 8] = raB[it];
      }
    } else {
      #pragma unroll
      for (int it = 0; it < 4; ++it) {
        int idx = tid + it * 256;
        int r = idx >> 3, c4 = idx & 7;
        uint2 st;
        st.x = f2bf_pk(raF[it].x, raF[it].y);
        st.y = f2bf_pk(raF[it].z, raF[it].w);
        *(uint2*)&sA[0][r * LDK + c4 * 4] = st;
      }
    }
  };

  int cur = 0;
  fetchB(0, 0);
  fetchA(0, 0);

  for (int kb = 0; kb < Ktot; kb += BKK) {
    if constexpr (!AASYNC) storeA();
#if USE_ASYNC
    __builtin_amdgcn_s_wait_asynccnt(0);         // async tiles for `cur` landed in LDS
#else
    *(uint4*)&sBh[cur][br * LDK + bc * 8] = rbh;
    *(uint4*)&sBg[cur][br * LDK + bc * 8] = rbg;
#endif
    __syncthreads();

    if (kb + BKK < Ktot) {
      fetchA(kb + BKK, cur ^ 1);
      fetchB(kb + BKK, cur ^ 1);                 // overlaps with WMMAs below
    }

    const int sab = AASYNC ? cur : 0;
    const int a0 = ((waveM * 2 + 0) * 16 + laneN) * LDK + half * 8;
    const int a1 = ((waveM * 2 + 1) * 16 + laneN) * LDK + half * 8;
    const int b0 = ((waveN * 2 + 0) * 16 + laneN) * LDK + half * 8;
    const int b1 = ((waveN * 2 + 1) * 16 + laneN) * LDK + half * 8;

    Frag fa0, fa1, fbh0, fbh1, fbg0, fbg1;
    fa0.u.lo  = *(const v4u*)&sA[sab][a0];  fa0.u.hi  = *(const v4u*)&sA[sab][a0 + 16];
    fa1.u.lo  = *(const v4u*)&sA[sab][a1];  fa1.u.hi  = *(const v4u*)&sA[sab][a1 + 16];
    fbh0.u.lo = *(const v4u*)&sBh[cur][b0]; fbh0.u.hi = *(const v4u*)&sBh[cur][b0 + 16];
    fbh1.u.lo = *(const v4u*)&sBh[cur][b1]; fbh1.u.hi = *(const v4u*)&sBh[cur][b1 + 16];
    fbg0.u.lo = *(const v4u*)&sBg[cur][b0]; fbg0.u.hi = *(const v4u*)&sBg[cur][b0 + 16];
    fbg1.u.lo = *(const v4u*)&sBg[cur][b1]; fbg1.u.hi = *(const v4u*)&sBg[cur][b1 + 16];

    ah00 = wmma_bf16(fa0, fbh0, ah00);
    ah01 = wmma_bf16(fa0, fbh1, ah01);
    ah10 = wmma_bf16(fa1, fbh0, ah10);
    ah11 = wmma_bf16(fa1, fbh1, ah11);
    ag00 = wmma_bf16(fa0, fbg0, ag00);
    ag01 = wmma_bf16(fa0, fbg1, ag01);
    ag10 = wmma_bf16(fa1, fbg0, ag10);
    ag11 = wmma_bf16(fa1, fbg1, ag11);

    __syncthreads();
    cur ^= 1;
  }

  // ---- epilogue ----
  const int n0g = nbase + waveN * 32 + laneN;
  const int n1g = n0g + 16;
  const float bhv0 = bh[n0g], bhv1 = bh[n1g];
  const float bgv0 = bg[n0g], bgv1 = bg[n1g];

  #pragma unroll
  for (int mi = 0; mi < 2; ++mi) {
    const v8f& h0 = mi ? ah10 : ah00;
    const v8f& h1 = mi ? ah11 : ah01;
    const v8f& g0 = mi ? ag10 : ag00;
    const v8f& g1 = mi ? ag11 : ag01;
    #pragma unroll
    for (int r = 0; r < 8; ++r) {
      int m = mbase + (waveM * 2 + mi) * 16 + half * 8 + r;
      float v0 = tanh_fast(h0[r] + bhv0) * sigmoid_fast(g0[r] + bgv0);
      float v1 = tanh_fast(h1[r] + bhv1) * sigmoid_fast(g1[r] + bgv1);
      size_t o0 = (size_t)m * HDIM + n0g;
      size_t o1 = (size_t)m * HDIM + n1g;
      if constexpr (OUTBF16) {
        unsigned short* o = (unsigned short*)outv;
        o[o0] = f2bf(v0);
        o[o1] = f2bf(v1);
      } else {
        float* o = (float*)outv;
        if (group != nullptr) {
          int g = group[m];
          float l0 = o[o0], l1 = o[o1];
          v0 = (g == 0) ? l0 : ((g == 1) ? v0 : 0.0f);
          v1 = (g == 0) ? l1 : ((g == 1) ? v1 : 0.0f);
        }
        o[o0] = v0;
        o[o1] = v1;
      }
    }
  }
}

// ---- one-time pre-passes ---------------------------------------------------

__global__ __launch_bounds__(256)
void transpose_w_kernel(const float* __restrict__ W, unsigned short* __restrict__ Wt, int K) {
  __shared__ __align__(16) unsigned short t[64 * 72];
  const int tid = threadIdx.x;
  const int kb = blockIdx.x * 64, nb = blockIdx.y * 64;
  #pragma unroll
  for (int it = 0; it < 4; ++it) {
    int idx = tid + it * 256;
    int kr = idx >> 4, c4 = idx & 15;
    const float4 w = *(const float4*)(W + (size_t)(kb + kr) * HDIM + nb + c4 * 4);
    int n = c4 * 4;
    t[(n + 0) * 72 + kr] = f2bf(w.x);
    t[(n + 1) * 72 + kr] = f2bf(w.y);
    t[(n + 2) * 72 + kr] = f2bf(w.z);
    t[(n + 3) * 72 + kr] = f2bf(w.w);
  }
  __syncthreads();
  #pragma unroll
  for (int it = 0; it < 2; ++it) {
    int idx = tid + it * 256;
    int nr = idx >> 3, c = idx & 7;
    *(uint4*)(Wt + (size_t)(nb + nr) * K + kb + c * 8) = *(const uint4*)&t[nr * 72 + c * 8];
  }
}

__global__ void cvt_bf16_kernel(const float* __restrict__ src, unsigned short* __restrict__ dst) {
  int i = blockIdx.x * blockDim.x + threadIdx.x;       // one float4 per thread
  const float4 a = *(const float4*)(src + (size_t)i * 4);
  uint2 b;
  b.x = f2bf_pk(a.x, a.y);
  b.y = f2bf_pk(a.z, a.w);
  *(uint2*)(dst + (size_t)i * 4) = b;
}

__global__ void mask_kernel(const int* __restrict__ group, int* __restrict__ mask, int n) {
  int i = blockIdx.x * blockDim.x + threadIdx.x;
  if (i < n) mask[i] = (group[i] == 2) ? 1 : 0;
}

// ---- host launch -----------------------------------------------------------

extern "C" void kernel_launch(void* const* d_in, const int* in_sizes, int n_in,
                              void* d_out, int out_size, void* d_ws, size_t ws_size,
                              hipStream_t stream) {
  (void)in_sizes; (void)n_in; (void)out_size; (void)ws_size;

  const float* node_hidden     = (const float*)d_in[0];
  const float* node_context    = (const float*)d_in[1];
  const float* label_embedding = (const float*)d_in[2];
  const float* left_embedding  = (const float*)d_in[3];
  const int*   group           = (const int*)d_in[4];
  const float* Wl1h = (const float*)d_in[5];  const float* bl1h = (const float*)d_in[6];
  const float* Wl1g = (const float*)d_in[7];  const float* bl1g = (const float*)d_in[8];
  const float* Wl2h = (const float*)d_in[9];  const float* bl2h = (const float*)d_in[10];
  const float* Wl2g = (const float*)d_in[11]; const float* bl2g = (const float*)d_in[12];
  const float* Wr1h = (const float*)d_in[13]; const float* br1h = (const float*)d_in[14];
  const float* Wr1g = (const float*)d_in[15]; const float* br1g = (const float*)d_in[16];
  const float* Wr2h = (const float*)d_in[17]; const float* br2h = (const float*)d_in[18];
  const float* Wr2g = (const float*)d_in[19]; const float* br2g = (const float*)d_in[20];

  const int Bn = 32768, H = 512;
  const size_t BH = (size_t)Bn * H;

  unsigned short* w = (unsigned short*)d_ws;
  unsigned short* tWl1h = w;  w += (size_t)3 * H * H;
  unsigned short* tWl1g = w;  w += (size_t)3 * H * H;
  unsigned short* tWr1h = w;  w += (size_t)3 * H * H;
  unsigned short* tWr1g = w;  w += (size_t)3 * H * H;
  unsigned short* tWl2h = w;  w += (size_t)H * H;
  unsigned short* tWl2g = w;  w += (size_t)H * H;
  unsigned short* tWr2h = w;  w += (size_t)2 * H * H;
  unsigned short* tWr2g = w;  w += (size_t)2 * H * H;
  unsigned short* left_bf = w; w += BH;
  unsigned short* h_l = w;     w += BH;
  unsigned short* h_r = w;     w += BH;

  float* children = (float*)d_out;
  int*   fin = (int*)d_out + BH;

  dim3 blk(256);

  transpose_w_kernel<<<dim3(24, 8), blk, 0, stream>>>(Wl1h, tWl1h, 3 * H);
  transpose_w_kernel<<<dim3(24, 8), blk, 0, stream>>>(Wl1g, tWl1g, 3 * H);
  transpose_w_kernel<<<dim3(24, 8), blk, 0, stream>>>(Wr1h, tWr1h, 3 * H);
  transpose_w_kernel<<<dim3(24, 8), blk, 0, stream>>>(Wr1g, tWr1g, 3 * H);
  transpose_w_kernel<<<dim3( 8, 8), blk, 0, stream>>>(Wl2h, tWl2h, H);
  transpose_w_kernel<<<dim3( 8, 8), blk, 0, stream>>>(Wl2g, tWl2g, H);
  transpose_w_kernel<<<dim3(16, 8), blk, 0, stream>>>(Wr2h, tWr2h, 2 * H);
  transpose_w_kernel<<<dim3(16, 8), blk, 0, stream>>>(Wr2g, tWr2g, 2 * H);
  cvt_bf16_kernel<<<(int)(BH / 4 / 256), blk, 0, stream>>>(left_embedding, left_bf);

  dim3 grid(H / BN, Bn / BM);   // (8, 256)

  gate_gemm<false, true><<<grid, blk, 0, stream>>>(node_hidden, node_context, label_embedding,
                                                   3 * H, tWl1h, bl1h, tWl1g, bl1g, h_l, nullptr);
  gate_gemm<false, true><<<grid, blk, 0, stream>>>(node_hidden, node_context, label_embedding,
                                                   3 * H, tWr1h, br1h, tWr1g, br1g, h_r, nullptr);
  gate_gemm<true, false><<<grid, blk, 0, stream>>>(h_l, nullptr, nullptr,
                                                   H, tWl2h, bl2h, tWl2g, bl2g, children, nullptr);
  gate_gemm<true, false><<<grid, blk, 0, stream>>>(h_r, left_bf, nullptr,
                                                   2 * H, tWr2h, br2h, tWr2g, br2g, children, group);

  mask_kernel<<<Bn / 256, blk, 0, stream>>>(group, fin, Bn);
}
